// stp_poolingnet_36378372997401
// MI455X (gfx1250) — matrix operand order repaired and checked by reading, not compiled, not run on hardware
//
#include <hip/hip_runtime.h>
#include <cstdint>

typedef __attribute__((ext_vector_type(16))) _Float16 v16h;
typedef __attribute__((ext_vector_type(8)))  float    v8f;

#define N_  1536
#define H_  64
#define E_  16
#define M_  128
#define T_  20
#define ITILES 96     // N/16
#define JC_ 32        // j-chunks
#define JLEN 48       // N / JC_

// ---------------------------------------------------------------------------
// Kernel 1: P[j][m] = h_j@W1h + u_j@W1e ;  Q[i][m] = u_i@W1e - be@W1e - b1
// stored as f16, permuted into the CDNA5 16-bit A-matrix per-lane layout:
//   chunk c = m>>5, km = m&31, khalf = (km>>3)&1, slot s = (km&7) + ((km&16)>>1)
//   linear offset = ((r*4 + c)*2 + khalf)*16 + s   (each lane's 16 halves contiguous)
// ---------------------------------------------------------------------------
__global__ __launch_bounds__(256) void precompute_pq(
    const float* __restrict__ hidden, const float* __restrict__ gt,
    const float* __restrict__ We, const float* __restrict__ be,
    const float* __restrict__ W1, const float* __restrict__ b1,
    _Float16* __restrict__ Pw, _Float16* __restrict__ Qw)
{
  int idx = blockIdx.x * 256 + threadIdx.x;
  if (idx >= N_ * M_) return;
  int r = idx >> 7, m = idx & 127;
  float e0 = gt[r * (T_ * 2) + (T_ - 1) * 2 + 0];
  float e1 = gt[r * (T_ * 2) + (T_ - 1) * 2 + 1];
  float accA = 0.f, accC = 0.f;
#pragma unroll
  for (int e = 0; e < E_; ++e) {
    float u = e0 * We[e] + e1 * We[E_ + e];
    float w = W1[(H_ + e) * M_ + m];
    accA += u * w;
    accC += be[e] * w;
  }
  float accP = accA;
#pragma unroll 8
  for (int h = 0; h < H_; ++h) accP += hidden[r * H_ + h] * W1[h * M_ + m];
  float Qv = accA - accC - b1[m];

  int c  = m >> 5;
  int km = m & 31;
  int kh = (km >> 3) & 1;
  int s  = (km & 7) + ((km & 16) >> 1);
  int off = ((r * 4 + c) * 2 + kh) * 16 + s;
  Pw[off] = (_Float16)accP;
  Qw[off] = (_Float16)Qv;
}

// ---------------------------------------------------------------------------
// Kernel 2: repack W2 (128x64 f32) into f16 WMMA B-matrix lane layout:
//   tile t (N cols 16t..16t+15), chunk c (K 32c..32c+31), lane L, slot s:
//   n = t*16 + (L&15),  k = c*32 + (L>>4)*8 + (s&7) + (s>>3)*16
// ---------------------------------------------------------------------------
__global__ __launch_bounds__(256) void repack_w2(
    const float* __restrict__ W2, _Float16* __restrict__ Bw)
{
  int idx = blockIdx.x * 256 + threadIdx.x;
  if (idx >= 4 * 4 * 32 * 16) return;
  int s = idx & 15;
  int L = (idx >> 4) & 31;
  int c = (idx >> 9) & 3;
  int t = idx >> 11;
  int n = t * 16 + (L & 15);
  int k = c * 32 + (L >> 4) * 8 + (s & 7) + ((s >> 3) << 4);
  Bw[idx] = (_Float16)W2[k * 64 + n];
}

// ---------------------------------------------------------------------------
// Kernel 3: per wave: 16-row i-tile x 32 output cols (2 WMMA N-tiles),
// loop j over a chunk; A = relu(P[j]-Q[i]) built in packed f16, 8 x
// v_wmma_f32_16x16x32_f16 per j (K=128 chained), running f32 max,
// reduced across chunks with global_atomic_max_num_f32 (non-returning).
// ---------------------------------------------------------------------------
__global__ __launch_bounds__(64) void pool_wmma(
    const _Float16* __restrict__ Pw, const _Float16* __restrict__ Qw,
    const _Float16* __restrict__ Bw, const float* __restrict__ b2,
    float* __restrict__ out)
{
  const int lane  = threadIdx.x & 31;
  const int wv    = threadIdx.x >> 5;
  const int it    = blockIdx.x % ITILES;
  const int jc    = blockIdx.x / ITILES;
  const int ibase = it * 16;
  const int khalf = lane >> 4;
  const int lrow  = lane & 15;

  // Q rows for this i-tile, register resident (A-layout, 4 chunks)
  v16h q[4];
#pragma unroll
  for (int c = 0; c < 4; ++c)
    q[c] = *(const v16h*)(Qw + ((ibase + lrow) * 4 + c) * 32 + khalf * 16);

  // W2 in B-layout, register resident: this wave's two N-tiles x 4 K-chunks
  v16h Bm[2][4];
#pragma unroll
  for (int tt = 0; tt < 2; ++tt) {
    int t = wv * 2 + tt;
#pragma unroll
    for (int c = 0; c < 4; ++c)
      Bm[tt][c] = *(const v16h*)(Bw + ((t * 4 + c) * 32 + lane) * 16);
  }

  v16h hzero;
#pragma unroll
  for (int s = 0; s < 16; ++s) hzero[s] = (_Float16)0.f;
  v8f zc, zmax0, zmax1;
#pragma unroll
  for (int v = 0; v < 8; ++v) {
    zc[v] = 0.f;
    zmax0[v] = -3.0e38f;
    zmax1[v] = -3.0e38f;
  }

  const int j0 = jc * JLEN, j1 = j0 + JLEN;
  for (int j = j0; j < j1; ++j) {
    const _Float16* prow = Pw + j * 128 + khalf * 16;
    __builtin_prefetch(prow + 128, 0, 1);   // global_prefetch_b8 of next P row
    v8f z0 = zc, z1 = zc;
#pragma unroll
    for (int c = 0; c < 4; ++c) {
      v16h p = *(const v16h*)(prow + c * 32);
      v16h a = __builtin_elementwise_max(p - q[c], hzero);  // relu(P-Q), pk f16
      z0 = __builtin_amdgcn_wmma_f32_16x16x32_f16(false, a, false, Bm[0][c],
                                                  (short)0, z0, false, false);
      z1 = __builtin_amdgcn_wmma_f32_16x16x32_f16(false, a, false, Bm[1][c],
                                                  (short)0, z1, false, false);
    }
    zmax0 = __builtin_elementwise_max(zmax0, z0);
    zmax1 = __builtin_elementwise_max(zmax1, z1);
  }

  // epilogue: out[i][n] = relu(max_j z + b2[n]); combine chunks via atomic max
#pragma unroll
  for (int tt = 0; tt < 2; ++tt) {
    const int t   = wv * 2 + tt;
    const int col = t * 16 + lrow;
    const float bias = b2[col];
    v8f zm = tt ? zmax1 : zmax0;
#pragma unroll
    for (int v = 0; v < 8; ++v) {
      float val = fmaxf(zm[v] + bias, 0.f);
      float* dst = out + (size_t)(ibase + khalf * 8 + v) * H_ + col;
      unsigned long long a64 = (unsigned long long)(uintptr_t)dst;
      asm volatile("global_atomic_max_num_f32 %0, %1, off"
                   :: "v"(a64), "v"(val) : "memory");
    }
  }
}

// ---------------------------------------------------------------------------
extern "C" void kernel_launch(void* const* d_in, const int* in_sizes, int n_in,
                              void* d_out, int out_size, void* d_ws, size_t ws_size,
                              hipStream_t stream)
{
  (void)in_sizes; (void)n_in; (void)ws_size;
  const float* hidden = (const float*)d_in[0];
  const float* gt     = (const float*)d_in[1];
  const float* We     = (const float*)d_in[2];
  const float* be     = (const float*)d_in[3];
  const float* W1     = (const float*)d_in[4];
  const float* b1     = (const float*)d_in[5];
  const float* W2     = (const float*)d_in[6];
  const float* b2     = (const float*)d_in[7];
  float* out = (float*)d_out;

  _Float16* Pw = (_Float16*)d_ws;          // N*128 halves
  _Float16* Qw = Pw + N_ * M_;             // N*128 halves
  _Float16* Bw = Qw + N_ * M_;             // 8192 halves

  hipMemsetAsync(d_out, 0, (size_t)out_size * sizeof(float), stream);
  precompute_pq<<<(N_ * M_ + 255) / 256, 256, 0, stream>>>(hidden, gt, We, be,
                                                           W1, b1, Pw, Qw);
  repack_w2<<<(8192 + 255) / 256, 256, 0, stream>>>(W2, Bw);
  pool_wmma<<<ITILES * JC_, 64, 0, stream>>>(Pw, Qw, Bw, b2, out);
}
